// ShortTermTemporalDecoder_48421461295830
// MI455X (gfx1250) — matrix operand order repaired and checked by reading, not compiled, not run on hardware
//
#include <hip/hip_runtime.h>

// ---------------------------------------------------------------------------
// ShortTermTemporalDecoder for MI455X (gfx1250), wave32 + WMMA bf16.
// ---------------------------------------------------------------------------

#define NQ_    40000
#define D_     256
#define DFF_   1024
#define NH_    8
#define NP_    8
#define HD_    32
#define BEVW_  200
#define BEVH_  200

typedef __bf16 bf16_t;
typedef __attribute__((ext_vector_type(16))) __bf16 v16bf;
typedef __attribute__((ext_vector_type(8)))  float  v8f;

// ---------------------------------------------------------------------------
// WMMA fragment loaders (contiguous k order, same for A and B)
// A (16x32 bf16): lanes 0-15 row M=lane, K=k0..k0+15;
//                 lanes 16-31 row M=lane-16, K=k0+16..k0+31.
// ---------------------------------------------------------------------------
__device__ __forceinline__ v16bf load_a_frag_bf16(const bf16_t* __restrict__ A,
                                                  int lda, int M0, int k0, int lane) {
    const int half = (lane >> 4) & 1;
    const int m    = lane & 15;
    const bf16_t* p = A + (size_t)(M0 + m) * lda + k0 + half * 16;
    v16bf a;
#pragma unroll
    for (int i = 0; i < 16; ++i) {
        a[i] = p[i];
    }
    return a;
}

// B (32x16 bf16, K x N): lanes 0-15 col N=lane, K=k0..k0+15;
//                        lanes 16-31 col N=lane-16, K=k0+16..k0+31.
// Bt is the pre-transposed weight (N rows, K contiguous) -> contiguous loads.
__device__ __forceinline__ v16bf load_b_frag(const bf16_t* __restrict__ Bt,
                                             int ldb, int N0, int k0, int lane) {
    const int n  = lane & 15;
    const int kb = k0 + ((lane >> 4) & 1) * 16;
    const bf16_t* p = Bt + (size_t)(N0 + n) * ldb + kb;
    v16bf b;
#pragma unroll
    for (int i = 0; i < 16; ++i) b[i] = p[i];
    return b;
}

// ---------------------------------------------------------------------------
// Elementwise f32 -> bf16 convert (4 elements/thread, vectorized).
// ---------------------------------------------------------------------------
__global__ void k_cvt(const float* __restrict__ X, bf16_t* __restrict__ Y, int n4) {
    int i = blockIdx.x * blockDim.x + threadIdx.x;
    if (i >= n4) return;
    const float4 v = ((const float4*)X)[i];
    bf16_t* y = Y + (size_t)i * 4;
    y[0] = (bf16_t)v.x; y[1] = (bf16_t)v.y; y[2] = (bf16_t)v.z; y[3] = (bf16_t)v.w;
}

// Weight transpose + bf16 convert:  Wt[n*K + k] = bf16(W[k*N + n])
__global__ void k_wt(const float* __restrict__ W, bf16_t* __restrict__ Wt,
                     int K, int N) {
    int i = blockIdx.x * blockDim.x + threadIdx.x;
    if (i >= N * K) return;
    int n = i / K, k = i - n * K;
    Wt[i] = (bf16_t)W[(size_t)k * N + n];
}

// ---------------------------------------------------------------------------
// Generic GEMM: C[M,N] = A[M,K] (bf16) x Bt[N,K] (bf16) + epilogue.
// One wave computes a 32x64 tile (2 M-subtiles x 4 N-subtiles, 8 acc);
// each B fragment feeds two WMMAs.
// ---------------------------------------------------------------------------
enum { EPI_BIAS = 0, EPI_BSHORT = 1 };

template <int EPI>
__global__ __launch_bounds__(256)
void k_gemm(const bf16_t* __restrict__ A, const bf16_t* __restrict__ Bt,
            const float* __restrict__ bias, const float* __restrict__ add0,
            float* __restrict__ C, bf16_t* __restrict__ Cbf,
            int M, int N, int K) {
    const int lane   = threadIdx.x & 31;
    const int wid    = blockIdx.x * (blockDim.x >> 5) + (threadIdx.x >> 5);
    const int ntiles = N >> 6;
    const int mt = wid / ntiles, nt = wid - mt * ntiles;
    const int M0 = mt * 32, N0 = nt * 64;
    if (M0 >= M) return;

    v8f acc0[4] = {};
    v8f acc1[4] = {};
    for (int k0 = 0; k0 < K; k0 += 32) {
        v16bf a0 = load_a_frag_bf16(A, K, M0,      k0, lane);
        v16bf a1 = load_a_frag_bf16(A, K, M0 + 16, k0, lane);
#pragma unroll
        for (int j = 0; j < 4; ++j) {
            v16bf b = load_b_frag(Bt, K, N0 + 16 * j, k0, lane);
            acc0[j] = __builtin_amdgcn_wmma_f32_16x16x32_bf16(
                false, a0, false, b, (short)0, acc0[j], false, false);
            acc1[j] = __builtin_amdgcn_wmma_f32_16x16x32_bf16(
                false, a1, false, b, (short)0, acc1[j], false, false);
        }
    }

    const int half = lane >> 4, n = lane & 15;
#pragma unroll
    for (int j = 0; j < 4; ++j) {
        const int col = N0 + 16 * j + n;
        const float bv = bias[col];
#pragma unroll
        for (int r = 0; r < 8; ++r) {
            const int row0 = M0 + r + 8 * half;
            const int row1 = row0 + 16;
            float v0 = acc0[j][r];
            float v1 = acc1[j][r];
            if (EPI == EPI_BSHORT) {
                v0 += 2.f * bv + 2.f * add0[(size_t)row0 * N + col];
                v1 += 2.f * bv + 2.f * add0[(size_t)row1 * N + col];
                Cbf[(size_t)row0 * N + col] = (bf16_t)v0;
                Cbf[(size_t)row1 * N + col] = (bf16_t)v1;
            } else {
                v0 += bv;
                v1 += bv;
            }
            C[(size_t)row0 * N + col] = v0;
            C[(size_t)row1 * N + col] = v1;
        }
    }
}

// ---------------------------------------------------------------------------
// Post: ref = sigmoid(q @ W_ref + b_ref); loc = ref + off/200; softmax(attn).
// One wave per query: coalesced dot + shfl reduction.
// ---------------------------------------------------------------------------
__global__ __launch_bounds__(256)
void k_post(const float* __restrict__ q, const float* __restrict__ Wref,
            const float* __restrict__ bref,
            const float* __restrict__ off_raw,
            float* __restrict__ attn,   // in: logits, out: weights
            float* __restrict__ loc, int nq) {
    const int lane = threadIdx.x & 31;
    const int qi   = blockIdx.x * (blockDim.x >> 5) + (threadIdx.x >> 5);
    if (qi >= nq) return;
    const float* qr = q + (size_t)qi * D_;
    float rx = 0.f, ry = 0.f;
#pragma unroll
    for (int t = 0; t < D_ / 32; ++t) {
        const int k = lane + 32 * t;
        const float v = qr[k];
        rx += v * Wref[2 * k];
        ry += v * Wref[2 * k + 1];
    }
#pragma unroll
    for (int off = 16; off > 0; off >>= 1) {
        rx += __shfl_xor(rx, off, 32);
        ry += __shfl_xor(ry, off, 32);
    }
    rx = 1.f / (1.f + __expf(-(rx + bref[0])));
    ry = 1.f / (1.f + __expf(-(ry + bref[1])));

    const float* off_q = off_raw + (size_t)qi * (NH_ * NP_ * 2);
    float* lq = loc + (size_t)qi * (NH_ * NP_ * 2);
    const float inv_n = 1.f / 200.f;
#pragma unroll
    for (int t = 0; t < 2; ++t) {
        const int j = lane + 32 * t;       // j in [0,64): (h,p) pair
        lq[2 * j]     = rx + off_q[2 * j] * inv_n;
        lq[2 * j + 1] = ry + off_q[2 * j + 1] * inv_n;
    }

    if (lane < NH_) {
        float* aq = attn + (size_t)qi * (NH_ * NP_) + lane * NP_;
        float m = -1e30f;
#pragma unroll
        for (int p = 0; p < NP_; ++p) m = fmaxf(m, aq[p]);
        float e[NP_], s = 0.f;
#pragma unroll
        for (int p = 0; p < NP_; ++p) { e[p] = __expf(aq[p] - m); s += e[p]; }
        const float inv = 1.f / s;
#pragma unroll
        for (int p = 0; p < NP_; ++p) aq[p] = e[p] * inv;
    }
}

// ---------------------------------------------------------------------------
// Deformable bilinear gather. One wave per (query, head); lane = channel d.
// S[q, h*32+d] = sum_{V,p,corner} w * val_V[(y*200+x)*256 + h*32 + d]
// (attend is linear in value, so both frames' maps share one weight.)
// Writes S directly as bf16 (next consumer is a WMMA GEMM).
// ---------------------------------------------------------------------------
__global__ __launch_bounds__(256)
void k_deform(const float* __restrict__ val0, const float* __restrict__ val1,
              const float* __restrict__ loc, const float* __restrict__ attnw,
              bf16_t* __restrict__ S, int nq) {
    const int lane = threadIdx.x & 31;
    const int wid  = blockIdx.x * (blockDim.x >> 5) + (threadIdx.x >> 5);
    const int qi = wid >> 3, h = wid & 7;
    if (qi >= nq) return;
    const float* lq = loc   + ((size_t)qi * NH_ + h) * (NP_ * 2);
    const float* aq = attnw + ((size_t)qi * NH_ + h) * NP_;
    float acc = 0.f;
#pragma unroll
    for (int p = 0; p < NP_; ++p) {
        float x = lq[2 * p]     * (float)BEVW_ - 0.5f;
        float y = lq[2 * p + 1] * (float)BEVH_ - 0.5f;
        float xf = floorf(x), yf = floorf(y);
        float lx = x - xf, ly = y - yf;
        int x0 = (int)xf, y0 = (int)yf;
        float aw = aq[p];
        float cw[4] = { (1.f - lx) * (1.f - ly), lx * (1.f - ly),
                        (1.f - lx) * ly,          lx * ly };
#pragma unroll
        for (int c = 0; c < 4; ++c) {
            int ix = x0 + (c & 1), iy = y0 + (c >> 1);
            if (ix < 0 || ix >= BEVW_ || iy < 0 || iy >= BEVH_) continue;
            size_t base = ((size_t)(iy * BEVW_ + ix)) * D_ + h * HD_ + lane;
            acc += aw * cw[c] * (val0[base] + val1[base]);
        }
    }
    S[(size_t)qi * D_ + h * HD_ + lane] = (bf16_t)acc;
}

// ---------------------------------------------------------------------------
// Fused FFN: out = Bshort + relu(Bshort@W1 + b1)@W2 + b2
// 4 waves/block; each wave owns 16 rows; hidden 16x1024 bf16 tile in LDS
// (32 KB/wave = 128 KB/block, under the 320 KB/WGP budget).
// ---------------------------------------------------------------------------
#define FFN_WAVES 4

__global__ __launch_bounds__(128)
void k_ffn(const bf16_t* __restrict__ Bs_bf, const float* __restrict__ Bshort,
           const bf16_t* __restrict__ Wt1, const float* __restrict__ b1,
           const bf16_t* __restrict__ Wt2, const float* __restrict__ b2,
           float* __restrict__ out, int M) {
    __shared__ bf16_t lds[FFN_WAVES][16][DFF_];
    const int lane = threadIdx.x & 31;
    const int w    = threadIdx.x >> 5;
    const int mt   = blockIdx.x * FFN_WAVES + w;
    const int M0   = mt * 16;
    if (M0 >= M) return;
    bf16_t* hrow = &lds[w][0][0];
    const int half = lane >> 4, n = lane & 15;

    // Phase 1: hidden = relu(Bshort @ W1 + b1) -> LDS (bf16)
    for (int nt = 0; nt < DFF_ / 16; ++nt) {
        v8f acc = {};
        for (int k0 = 0; k0 < D_; k0 += 32) {
            v16bf a = load_a_frag_bf16(Bs_bf, D_, M0, k0, lane);
            v16bf b = load_b_frag(Wt1, D_, nt * 16, k0, lane);
            acc = __builtin_amdgcn_wmma_f32_16x16x32_bf16(
                false, a, false, b, (short)0, acc, false, false);
        }
        const int col = nt * 16 + n;
        const float bv = b1[col];
#pragma unroll
        for (int r = 0; r < 8; ++r) {
            float v = acc[r] + bv;
            v = v > 0.f ? v : 0.f;
            hrow[(r + 8 * half) * DFF_ + col] = (bf16_t)v;
        }
    }
    __syncthreads();

    // Phase 2: out = hidden @ W2 + b2 + Bshort
    for (int nt = 0; nt < D_ / 16; ++nt) {
        v8f acc = {};
        for (int k0 = 0; k0 < DFF_; k0 += 32) {
            v16bf a = load_a_frag_bf16(hrow, DFF_, 0, k0, lane);
            v16bf b = load_b_frag(Wt2, DFF_, nt * 16, k0, lane);
            acc = __builtin_amdgcn_wmma_f32_16x16x32_bf16(
                false, a, false, b, (short)0, acc, false, false);
        }
        const int col = nt * 16 + n;
        const float bv = b2[col];
#pragma unroll
        for (int r = 0; r < 8; ++r) {
            const int row = M0 + r + 8 * half;
            out[(size_t)row * D_ + col] =
                acc[r] + bv + Bshort[(size_t)row * D_ + col];
        }
    }
}

// ---------------------------------------------------------------------------
// Host-side orchestration
// ---------------------------------------------------------------------------
static inline int gemm_blocks(int M, int N) {
    int waves = ((M + 31) / 32) * (N / 64);
    return (waves + 7) / 8;
}

extern "C" void kernel_launch(void* const* d_in, const int* in_sizes, int n_in,
                              void* d_out, int out_size, void* d_ws, size_t ws_size,
                              hipStream_t stream) {
    (void)in_sizes; (void)n_in; (void)out_size; (void)ws_size;
    const float* B_adj0 = (const float*)d_in[0];
    const float* B_adj1 = (const float*)d_in[1];
    const float* qemb   = (const float*)d_in[2];
    const float* W_ref  = (const float*)d_in[3];
    const float* b_ref  = (const float*)d_in[4];
    const float* W_off  = (const float*)d_in[5];
    const float* b_off  = (const float*)d_in[6];
    const float* W_attn = (const float*)d_in[7];
    const float* b_attn = (const float*)d_in[8];
    const float* W_val  = (const float*)d_in[9];
    const float* b_val  = (const float*)d_in[10];
    const float* W_out  = (const float*)d_in[11];
    const float* b_out  = (const float*)d_in[12];
    const float* W_ffn1 = (const float*)d_in[13];
    const float* b_ffn1 = (const float*)d_in[14];
    const float* W_ffn2 = (const float*)d_in[15];
    const float* b_ffn2 = (const float*)d_in[16];
    float* out = (float*)d_out;

    char* ws = (char*)d_ws;
    size_t o = 0;
    bf16_t* wt_off  = (bf16_t*)(ws + o); o += (size_t)128 * 256 * 2;
    bf16_t* wt_attn = (bf16_t*)(ws + o); o += (size_t)64 * 256 * 2;
    bf16_t* wt_val  = (bf16_t*)(ws + o); o += (size_t)256 * 256 * 2;
    bf16_t* wt_out  = (bf16_t*)(ws + o); o += (size_t)256 * 256 * 2;
    bf16_t* wt_f1   = (bf16_t*)(ws + o); o += (size_t)1024 * 256 * 2;
    bf16_t* wt_f2   = (bf16_t*)(ws + o); o += (size_t)256 * 1024 * 2;
    bf16_t* q_bf    = (bf16_t*)(ws + o); o += (size_t)NQ_ * D_ * 2;
    bf16_t* a0_bf   = (bf16_t*)(ws + o); o += (size_t)NQ_ * D_ * 2;
    bf16_t* a1_bf   = (bf16_t*)(ws + o); o += (size_t)NQ_ * D_ * 2;
    bf16_t* S_bf    = (bf16_t*)(ws + o); o += (size_t)NQ_ * D_ * 2;
    bf16_t* Bs_bf   = (bf16_t*)(ws + o); o += (size_t)NQ_ * D_ * 2;
    float* off_raw  = (float*)(ws + o);  o += (size_t)NQ_ * 128 * 4;
    float* attn     = (float*)(ws + o);  o += (size_t)NQ_ * 64 * 4;
    float* loc      = (float*)(ws + o);  o += (size_t)NQ_ * 128 * 4;
    float* val0     = (float*)(ws + o);  o += (size_t)NQ_ * D_ * 4;
    float* val1     = (float*)(ws + o);  o += (size_t)NQ_ * D_ * 4;
    float* Bshort   = (float*)(ws + o);  o += (size_t)NQ_ * D_ * 4;

    const int ND4 = NQ_ * D_ / 4;

    // 1) Weight transpose + activations -> bf16
    k_wt<<<(128 * 256 + 255) / 256, 256, 0, stream>>>(W_off, wt_off, 256, 128);
    k_wt<<<(64 * 256 + 255) / 256, 256, 0, stream>>>(W_attn, wt_attn, 256, 64);
    k_wt<<<(256 * 256 + 255) / 256, 256, 0, stream>>>(W_val, wt_val, 256, 256);
    k_wt<<<(256 * 256 + 255) / 256, 256, 0, stream>>>(W_out, wt_out, 256, 256);
    k_wt<<<(1024 * 256 + 255) / 256, 256, 0, stream>>>(W_ffn1, wt_f1, 256, 1024);
    k_wt<<<(256 * 1024 + 255) / 256, 256, 0, stream>>>(W_ffn2, wt_f2, 1024, 256);
    k_cvt<<<(ND4 + 255) / 256, 256, 0, stream>>>(qemb, q_bf, ND4);
    k_cvt<<<(ND4 + 255) / 256, 256, 0, stream>>>(B_adj0, a0_bf, ND4);
    k_cvt<<<(ND4 + 255) / 256, 256, 0, stream>>>(B_adj1, a1_bf, ND4);

    // 2) Query-side projections (WMMA)
    k_gemm<EPI_BIAS><<<gemm_blocks(NQ_, 128), 256, 0, stream>>>(
        q_bf, wt_off, b_off, nullptr, off_raw, nullptr, NQ_, 128, 256);
    k_gemm<EPI_BIAS><<<gemm_blocks(NQ_, 64), 256, 0, stream>>>(
        q_bf, wt_attn, b_attn, nullptr, attn, nullptr, NQ_, 64, 256);

    // 3) Value projections for both frames (WMMA)
    k_gemm<EPI_BIAS><<<gemm_blocks(NQ_, 256), 256, 0, stream>>>(
        a0_bf, wt_val, b_val, nullptr, val0, nullptr, NQ_, 256, 256);
    k_gemm<EPI_BIAS><<<gemm_blocks(NQ_, 256), 256, 0, stream>>>(
        a1_bf, wt_val, b_val, nullptr, val1, nullptr, NQ_, 256, 256);

    // 4) sigmoid(ref), sampling locations, softmax(attn)
    k_post<<<(NQ_ * 32 + 255) / 256, 256, 0, stream>>>(
        qemb, W_ref, b_ref, off_raw, attn, loc, NQ_);

    // 5) Deformable gather, both frames summed (linear in value) -> bf16
    k_deform<<<(NQ_ * NH_) / 8, 256, 0, stream>>>(val0, val1, loc, attn, S_bf, NQ_);

    // 6) Output projection + residuals: Bshort = S@W_out + 2*b_out + 2*q
    k_gemm<EPI_BSHORT><<<gemm_blocks(NQ_, 256), 256, 0, stream>>>(
        S_bf, wt_out, b_out, qemb, Bshort, Bs_bf, NQ_, 256, 256);

    // 7) Fused FFN with residual -> d_out
    k_ffn<<<(NQ_ / 16 + FFN_WAVES - 1) / FFN_WAVES, 32 * FFN_WAVES, 0, stream>>>(
        Bs_bf, Bshort, wt_f1, b_ffn1, wt_f2, b_ffn2, out, NQ_);
}
